// SelfAttentionLayer_33878702030958
// MI455X (gfx1250) — matrix operand order repaired and checked by reading, not compiled, not run on hardware
//
#include <hip/hip_runtime.h>
#include <hip/hip_bf16.h>

// ---------------------------------------------------------------------------
// MI455X (gfx1250) self-attention layer.
// B=2, N=2048, C=1024, H=16, hd=64.  All matrix math on V_WMMA_F32_16X16X32_BF16
// (bf16 inputs, f32 accumulation).  wave32 throughout.  Attention stages K/V
// tiles through LDS with double-buffered global_load_async_to_lds_b128.
// ---------------------------------------------------------------------------

typedef __bf16 bf16_t;
typedef __attribute__((ext_vector_type(16))) __bf16 v16bf;
typedef __attribute__((ext_vector_type(8)))  __bf16 v8bf;
typedef __attribute__((ext_vector_type(4)))  __bf16 v4bf;
typedef __attribute__((ext_vector_type(8)))  float  v8f;

__device__ inline v16bf cat8(v8bf lo, v8bf hi) {
  return __builtin_shufflevector(lo, hi, 0,1,2,3,4,5,6,7,8,9,10,11,12,13,14,15);
}

// A-fragment (16x32 bf16, row-major source, lane = M row):
//   lanes 0-15: K = kb+0..7 and 16+kb..23 with kb=0; lanes 16-31: kb=8
__device__ inline v16bf load_a_frag(const bf16_t* base, int lda, int row0, int k0) {
  int lane = threadIdx.x & 31;
  int m  = lane & 15;
  int kb = (lane >> 4) << 3;                       // 0 or 8
  const bf16_t* p = base + (size_t)(row0 + m) * lda + k0 + kb;
  v8bf lo = *(const v8bf*)p;                       // K = kb .. kb+7
  v8bf hi = *(const v8bf*)(p + 16);                // K = 16+kb .. 16+kb+7
  return cat8(lo, hi);
}

// B-fragment (32x16 bf16) from a PRE-TRANSPOSED matrix BT (N x K, row-major):
//   lane = N column; lanes 0-15: K = kb+0..7 and 16+kb..23 with kb=0;
//   lanes 16-31: kb=8 (same K order as load_a_frag).
__device__ inline v16bf load_bT_frag(const bf16_t* bt, int ldb, int col0, int k0) {
  int lane = threadIdx.x & 31;
  int n  = lane & 15;
  int kb = (lane >> 4) << 3;                       // 0 or 8
  const bf16_t* p = bt + (size_t)(col0 + n) * ldb + k0 + kb;
  v8bf lo = *(const v8bf*)p;
  v8bf hi = *(const v8bf*)(p + 16);
  return cat8(lo, hi);
}

__device__ inline v8f wmma_bf16(v16bf a, v16bf b, v8f c) {
  return __builtin_amdgcn_wmma_f32_16x16x32_bf16(false, a, false, b, (short)0, c,
                                                 false, false);
}

// xor-shuffle across lanes via ds_swizzle (group-of-32 mode, and_mask=0x1f).
#define SWZ_XOR(x, mask) \
  __int_as_float(__builtin_amdgcn_ds_swizzle(__float_as_int(x), ((mask) << 10) | 0x1f))

// Async global->LDS 16B copy (CDNA5, tracked by ASYNCcnt).
// VDST VGPR holds the LDS byte address (low 32 bits of the flat shared addr);
// VADDR pair holds the 64-bit global address.
__device__ inline void async_copy_b128(void* lds, const void* gptr) {
  unsigned l = (unsigned)(size_t)lds;
  unsigned long long g = (unsigned long long)(size_t)gptr;
  asm volatile("global_load_async_to_lds_b128 %0, %1, off"
               :: "v"(l), "v"(g) : "memory");
}
__device__ inline void wait_async0() {
  asm volatile("s_wait_asynccnt 0x0" ::: "memory");
}

// ---------------------------------------------------------------------------
// Elementwise converters
// ---------------------------------------------------------------------------
__global__ void cast_f32_bf16_kernel(const float* __restrict__ src,
                                     bf16_t* __restrict__ dst, int n4) {
  int id = blockIdx.x * blockDim.x + threadIdx.x;
  if (id >= n4) return;
  float4 v = ((const float4*)src)[id];
  v4bf o = {(bf16_t)v.x, (bf16_t)v.y, (bf16_t)v.z, (bf16_t)v.w};
  ((v4bf*)dst)[id] = o;
}

// W (K x N f32, row-major) -> WT (N x K bf16, row-major)
__global__ void transpose_cast_kernel(const float* __restrict__ W,
                                      bf16_t* __restrict__ WT, int K, int N) {
  int id = blockIdx.x * blockDim.x + threadIdx.x;
  if (id >= K * N) return;
  int k = id % K;
  int n = id / K;
  WT[id] = (bf16_t)W[(size_t)k * N + n];
}

// ---------------------------------------------------------------------------
// GEMM: C[M x N] (f32, +optional bias) = A[M x K] (bf16) * BT[N x K]^T (bf16)
// 256 threads = 8 waves (2m x 4n); block tile 128x256; wave tile 64x64
// (4x4 WMMA accumulators -> 16 WMMAs per 32-deep K step).
// Weights/activations are L2-resident (192 MB), so fragments load straight
// from global with per-lane-contiguous 16B requests.
// ---------------------------------------------------------------------------
__global__ __launch_bounds__(256) void gemm_bf16_kernel(
    const bf16_t* __restrict__ A, const bf16_t* __restrict__ BT,
    float* __restrict__ C, const float* __restrict__ bias,
    int M, int N, int K) {
  int w    = threadIdx.x >> 5;
  int lane = threadIdx.x & 31;
  int wm = w >> 2, wn = w & 3;
  int m0 = blockIdx.y * 128 + wm * 64;
  int n0 = blockIdx.x * 256 + wn * 64;
  (void)M;

  v8f zero = {};
  v8f acc[4][4];
#pragma unroll
  for (int i = 0; i < 4; ++i)
#pragma unroll
    for (int j = 0; j < 4; ++j) acc[i][j] = zero;

  for (int k0 = 0; k0 < K; k0 += 32) {
    v16bf af[4];
#pragma unroll
    for (int i = 0; i < 4; ++i) af[i] = load_a_frag(A, K, m0 + 16 * i, k0);
    v16bf bfr[4];
#pragma unroll
    for (int j = 0; j < 4; ++j) bfr[j] = load_bT_frag(BT, K, n0 + 16 * j, k0);
#pragma unroll
    for (int i = 0; i < 4; ++i)
#pragma unroll
      for (int j = 0; j < 4; ++j) acc[i][j] = wmma_bf16(af[i], bfr[j], acc[i][j]);
  }

  // C/D layout: lane holds column (lane&15); VGPR r -> row r (+8 for lanes>=16)
  int n  = lane & 15;
  int mh = (lane >> 4) << 3;
#pragma unroll
  for (int i = 0; i < 4; ++i) {
#pragma unroll
    for (int j = 0; j < 4; ++j) {
      int col = n0 + 16 * j + n;
      float bv = bias ? bias[col] : 0.0f;
#pragma unroll
      for (int r = 0; r < 8; ++r)
        C[(size_t)(m0 + 16 * i + mh + r) * N + col] = acc[i][j][r] + bv;
    }
  }
}

// ---------------------------------------------------------------------------
// RMS-norm (f32) on q,k per 64-wide head vector + repack to bf16:
//   qn, kn : [B][H][N][64]   vt : [B][H][64][N] (transposed for P*V B-frags)
// ---------------------------------------------------------------------------
__global__ void rms_repack_kernel(const float* __restrict__ qkv,
                                  const float* __restrict__ q_scale,
                                  const float* __restrict__ k_scale,
                                  bf16_t* __restrict__ qn, bf16_t* __restrict__ kn,
                                  bf16_t* __restrict__ vt) {
  int id  = blockIdx.x * blockDim.x + threadIdx.x;   // 3 * 65536
  int t   = id >> 16;                                // 0=q, 1=k, 2=v
  int rem = id & 65535;
  int b = rem >> 15;
  int h = (rem >> 11) & 15;
  int n = rem & 2047;

  const float* src = qkv + (size_t)(b * 2048 + n) * 3072 + t * 1024 + h * 64;
  float x[64];
  float ss = 0.f;
#pragma unroll
  for (int c = 0; c < 64; c += 4) {
    float4 v = *(const float4*)(src + c);
    x[c] = v.x; x[c + 1] = v.y; x[c + 2] = v.z; x[c + 3] = v.w;
    ss += v.x * v.x + v.y * v.y + v.z * v.z + v.w * v.w;
  }
  if (t == 2) {
    bf16_t* d = vt + (size_t)(b * 16 + h) * 64 * 2048 + n;
#pragma unroll
    for (int c = 0; c < 64; ++c) d[(size_t)c * 2048] = (bf16_t)x[c];
  } else {
    float rms = sqrtf(ss * 0.015625f);               // ||x|| / sqrt(64)
    float inv = 1.0f / (rms + 1e-8f);
    const float* sc = (t == 0) ? q_scale : k_scale;
    bf16_t* d = ((t == 0) ? qn : kn) + ((size_t)(b * 16 + h) * 2048 + n) * 64;
#pragma unroll
    for (int c = 0; c < 64; ++c) d[c] = (bf16_t)(sc[c] * x[c] * inv);
  }
}

// ---------------------------------------------------------------------------
// Flash attention, block-cooperative: all 8 waves of a block share one (b,h)
// and one double-buffered LDS K/V tile streamed in with async-to-LDS copies.
//   sK: [32 keys][72] bf16 (144B rows -> conflict-free ds_load_b128 B-frags)
//   sV: [64 hd ][40] bf16 ( 80B rows -> conflict-free)
//   sP: per-wave 16x32 bounce tile (80B rows) for C-layout -> A-layout.
// Each wave owns 16 query rows; 8 WMMAs per 32-key step.
// ---------------------------------------------------------------------------
__global__ __launch_bounds__(256) void attn_kernel(
    const bf16_t* __restrict__ qn, const bf16_t* __restrict__ kn,
    const bf16_t* __restrict__ vt, bf16_t* __restrict__ attn_out) {
  __shared__ __align__(16) bf16_t sK[2][32 * 72];
  __shared__ __align__(16) bf16_t sV[2][64 * 40];
  __shared__ __align__(16) bf16_t sP[8][16 * 40];

  int tid  = threadIdx.x;
  int w    = tid >> 5;
  int lane = tid & 31;
  int bh   = blockIdx.x >> 4;         // 0..31
  int mblk = blockIdx.x & 15;
  int m0   = (mblk * 8 + w) * 16;     // this wave's 16 query rows
  int b = bh >> 4, h = bh & 15;

  const bf16_t* qbase = qn + ((size_t)bh * 2048 + m0) * 64;
  const bf16_t* kbase = kn + (size_t)bh * 2048 * 64;
  const bf16_t* vbase = vt + (size_t)bh * 64 * 2048;

  // each thread stages one 16B chunk of K and one of V per tile
  int kRow = tid >> 3, kPart = tid & 7;   // 32 rows x 8 chunks (128B/row)
  int vRow = tid >> 2, vPart = tid & 3;   // 64 rows x 4 chunks ( 64B/row)
  auto stage = [&](int buf, int j0) {
    async_copy_b128(&sK[buf][kRow * 72 + kPart * 8],
                    kbase + (size_t)(j0 + kRow) * 64 + kPart * 8);
    async_copy_b128(&sV[buf][vRow * 40 + vPart * 8],
                    vbase + (size_t)vRow * 2048 + j0 + vPart * 8);
  };

  v16bf qa0 = load_a_frag(qbase, 64, 0, 0);
  v16bf qa1 = load_a_frag(qbase, 64, 0, 32);

  v8f zero = {};
  v8f o0 = zero, o1 = zero, o2 = zero, o3 = zero;
  float rm[8], rl[8];
#pragma unroll
  for (int r = 0; r < 8; ++r) { rm[r] = -3.0e38f; rl[r] = 0.f; }

  bf16_t* lp = sP[w];
  const int LROW = 40;
  int ncol = lane & 15;
  int half = lane >> 4;
  int mh   = half << 3;
  int kb2  = half << 3;

  stage(0, 0);
  wait_async0();
  __syncthreads();

  int buf = 0;
  for (int j0 = 0; j0 < 2048; j0 += 32) {
    if (j0 + 32 < 2048) stage(buf ^ 1, j0 + 32);   // prefetch next tile

    // S tiles (16 queries x 16 keys each), K-dim = hd = 64 -> 2 WMMAs each
    v8f s0 = zero, s1 = zero;
    s0 = wmma_bf16(qa0, load_bT_frag(sK[buf], 72, 0,   0), s0);
    s0 = wmma_bf16(qa1, load_bT_frag(sK[buf], 72, 0,  32), s0);
    s1 = wmma_bf16(qa0, load_bT_frag(sK[buf], 72, 16,  0), s1);
    s1 = wmma_bf16(qa1, load_bT_frag(sK[buf], 72, 16, 32), s1);

    float alpha[8], mn[8];
#pragma unroll
    for (int r = 0; r < 8; ++r) {
      float a = s0[r] * 0.125f;                    // hd^-0.5
      float c = s1[r] * 0.125f;
      s0[r] = a; s1[r] = c;
      float mx = fmaxf(a, c);
      mx = fmaxf(mx, SWZ_XOR(mx, 1));
      mx = fmaxf(mx, SWZ_XOR(mx, 2));
      mx = fmaxf(mx, SWZ_XOR(mx, 4));
      mx = fmaxf(mx, SWZ_XOR(mx, 8));
      mn[r]    = fmaxf(rm[r], mx);
      alpha[r] = __expf(rm[r] - mn[r]);
      rm[r]    = mn[r];
    }
#pragma unroll
    for (int r = 0; r < 8; ++r) {
      float p0 = __expf(s0[r] - mn[r]);
      float p1 = __expf(s1[r] - mn[r]);
      float ps = p0 + p1;
      ps += SWZ_XOR(ps, 1);
      ps += SWZ_XOR(ps, 2);
      ps += SWZ_XOR(ps, 4);
      ps += SWZ_XOR(ps, 8);
      rl[r] = rl[r] * alpha[r] + ps;
      lp[(r + mh) * LROW + ncol]      = (bf16_t)p0; // C-layout -> LDS
      lp[(r + mh) * LROW + 16 + ncol] = (bf16_t)p1;
      o0[r] *= alpha[r];
      o1[r] *= alpha[r];
      o2[r] *= alpha[r];
      o3[r] *= alpha[r];
    }
    asm volatile("s_wait_dscnt 0" ::: "memory");   // P stores visible (per-wave)

    // re-load P as A-fragment (16 queries x 32 keys)
    v8bf plo = *(const v8bf*)(lp + ncol * LROW + kb2);
    v8bf phi = *(const v8bf*)(lp + ncol * LROW + 16 + kb2);
    v16bf pA = cat8(plo, phi);

    // O += P * V : K = 32 keys, N = 64 hd split into 4 tiles
    o0 = wmma_bf16(pA, load_bT_frag(sV[buf], 40,  0, 0), o0);
    o1 = wmma_bf16(pA, load_bT_frag(sV[buf], 40, 16, 0), o1);
    o2 = wmma_bf16(pA, load_bT_frag(sV[buf], 40, 32, 0), o2);
    o3 = wmma_bf16(pA, load_bT_frag(sV[buf], 40, 48, 0), o3);

    wait_async0();                                  // next tile landed
    __syncthreads();                                // ..for every wave
    buf ^= 1;
  }

  // epilogue: normalize and scatter to [B][N][C] bf16 for the proj GEMM
#pragma unroll
  for (int r = 0; r < 8; ++r) {
    float inv = 1.0f / rl[r];
    size_t base = (size_t)(b * 2048 + m0 + mh + r) * 1024 + h * 64 + ncol;
    attn_out[base +  0] = (bf16_t)(o0[r] * inv);
    attn_out[base + 16] = (bf16_t)(o1[r] * inv);
    attn_out[base + 32] = (bf16_t)(o2[r] * inv);
    attn_out[base + 48] = (bf16_t)(o3[r] * inv);
  }
}

// ---------------------------------------------------------------------------
// Orchestration
// ---------------------------------------------------------------------------
extern "C" void kernel_launch(void* const* d_in, const int* in_sizes, int n_in,
                              void* d_out, int out_size, void* d_ws, size_t ws_size,
                              hipStream_t stream) {
  (void)in_sizes; (void)n_in; (void)out_size; (void)ws_size;
  const float* x      = (const float*)d_in[0];   // [2,2048,1024]
  const float* Wqkv   = (const float*)d_in[1];   // [1024,3072]
  const float* Wproj  = (const float*)d_in[2];   // [1024,1024]
  const float* bproj  = (const float*)d_in[3];   // [1024]
  const float* qscale = (const float*)d_in[4];   // [64]
  const float* kscale = (const float*)d_in[5];   // [64]
  float* out = (float*)d_out;                    // [2,2048,1024] f32

  char* ws = (char*)d_ws;
  const size_t OFF_XB     = 0;                                     //  8 MiB
  const size_t OFF_WQKVT  = OFF_XB     + (size_t)4096 * 1024 * 2;  //  6 MiB
  const size_t OFF_WPROJT = OFF_WQKVT  + (size_t)3072 * 1024 * 2;  //  2 MiB
  const size_t OFF_QKV    = OFF_WPROJT + (size_t)1024 * 1024 * 2;  // 48 MiB f32
  const size_t OFF_QN     = OFF_QKV    + (size_t)4096 * 3072 * 4;  //  8 MiB
  const size_t OFF_KN     = OFF_QN     + (size_t)4194304 * 2;      //  8 MiB
  const size_t OFF_VT     = OFF_KN     + (size_t)4194304 * 2;      //  8 MiB
  const size_t OFF_ATT    = OFF_VT     + (size_t)4194304 * 2;      //  8 MiB

  bf16_t* xb     = (bf16_t*)(ws + OFF_XB);
  bf16_t* wqkvT  = (bf16_t*)(ws + OFF_WQKVT);
  bf16_t* wprojT = (bf16_t*)(ws + OFF_WPROJT);
  float*  qkvf   = (float*) (ws + OFF_QKV);
  bf16_t* qnb    = (bf16_t*)(ws + OFF_QN);
  bf16_t* knb    = (bf16_t*)(ws + OFF_KN);
  bf16_t* vtb    = (bf16_t*)(ws + OFF_VT);
  bf16_t* attb   = (bf16_t*)(ws + OFF_ATT);

  // 1) precision/layout prep
  cast_f32_bf16_kernel<<<4096, 256, 0, stream>>>(x, xb, (4096 * 1024) / 4);
  transpose_cast_kernel<<<(3072 * 1024) / 256, 256, 0, stream>>>(Wqkv, wqkvT, 1024, 3072);
  transpose_cast_kernel<<<(1024 * 1024) / 256, 256, 0, stream>>>(Wproj, wprojT, 1024, 1024);

  // 2) qkv = x @ Wqkv  (f32 output for exact-ish RMS norm)
  gemm_bf16_kernel<<<dim3(3072 / 256, 4096 / 128), 256, 0, stream>>>(
      xb, wqkvT, qkvf, nullptr, 4096, 3072, 1024);

  // 3) RMS norm q,k + repack q/k/v for attention
  rms_repack_kernel<<<768, 256, 0, stream>>>(qkvf, qscale, kscale, qnb, knb, vtb);

  // 4) flash attention (32 bh-pairs x 16 blocks; 8 waves share LDS K/V tiles)
  attn_kernel<<<512, 256, 0, stream>>>(qnb, knb, vtb, attb);

  // 5) out = attn @ Wproj + bproj
  gemm_bf16_kernel<<<dim3(1024 / 256, 4096 / 128), 256, 0, stream>>>(
      attb, wprojT, out, bproj, 4096, 1024, 1024);
}
